// ClippedGRU_65111704207493
// MI455X (gfx1250) — compile-verified, matrix operand order checked by reading, not run on hardware
//
#include <hip/hip_runtime.h>

// ---------------------------------------------------------------------------
// Bidirectional clipped GRU for MI455X (gfx1250, wave32, WMMA 16x16x32 f16)
//   SEQ=512, BATCH=128, IN=512, HID=512
//   k_cvt    : f32->f16 conversions (x, w_ih, w_hh), h0 seed, barrier reset
//   k_xgemm  : xg = x @ w_ih^T  (both dirs), massively parallel WMMA GEMM
//   k_recur  : persistent per-direction recurrence; w_hh slice staged into
//              LDS with TENSOR_LOAD_TO_LDS (TDM), per-step WMMA hg = h@w_hh^T,
//              gate elementwise in f32, global atomic step barrier.
// ---------------------------------------------------------------------------

#define T_    512
#define B_    128
#define I_    512
#define H_    512
#define G_    (3 * H_)

typedef _Float16 v16h  __attribute__((ext_vector_type(16)));
typedef _Float16 half8 __attribute__((ext_vector_type(8)));
typedef float    v8f   __attribute__((ext_vector_type(8)));
typedef unsigned int uint4v __attribute__((ext_vector_type(4)));
typedef int      int8v  __attribute__((ext_vector_type(8)));
typedef int      int4v  __attribute__((ext_vector_type(4)));

// ---- workspace layout (bytes) ---------------------------------------------
static constexpr size_t XG_ELEMS  = (size_t)T_ * B_ * G_;
static constexpr size_t XF_ELEMS  = (size_t)T_ * B_ * I_;
static constexpr size_t WEL       = (size_t)G_ * I_;
static constexpr size_t OFF_XG_FW = 0;
static constexpr size_t OFF_XG_BW = OFF_XG_FW + XG_ELEMS * 4;
static constexpr size_t OFF_XF16  = OFF_XG_BW + XG_ELEMS * 4;
static constexpr size_t OFF_WIHF  = OFF_XF16  + XF_ELEMS * 2;
static constexpr size_t OFF_WIHB  = OFF_WIHF  + WEL * 2;
static constexpr size_t OFF_WHHF  = OFF_WIHB  + WEL * 2;
static constexpr size_t OFF_WHHB  = OFF_WHHF  + WEL * 2;
static constexpr size_t OFF_HF16  = OFF_WHHB  + WEL * 2;   // [2 dir][2 par][B][H] f16
static constexpr size_t OFF_HF32  = OFF_HF16  + (size_t)2 * 2 * B_ * H_ * 2;
static constexpr size_t OFF_BAR   = OFF_HF32  + (size_t)2 * 2 * B_ * H_ * 4;

// ---- WMMA helpers ----------------------------------------------------------
// 16-bit A/B fragment, 16x32 (MxK): lane half selects K-half, two contiguous
// 8-halfword runs per lane: [8*half .. +7], [16+8*half .. +7].
__device__ __forceinline__ v16h load_frag(const _Float16* base, int lhalf) {
  half8 r0 = *(const half8*)(base + 8 * lhalf);
  half8 r1 = *(const half8*)(base + 16 + 8 * lhalf);
  v16h f;
#pragma unroll
  for (int i = 0; i < 8; ++i) { f[i] = r0[i]; f[i + 8] = r1[i]; }
  return f;
}

__device__ __forceinline__ v8f wmma16(v16h a, v16h b, v8f c) {
  return __builtin_amdgcn_wmma_f32_16x16x32_f16(
      false, a, false, b, (short)0, c, false, false);
}

__device__ __forceinline__ float sigmoid_f(float x) {
  return 1.0f / (1.0f + __expf(-x));
}
__device__ __forceinline__ float tanh_f(float x) {
  float e2 = __expf(2.0f * x);
  return 1.0f - 2.0f / (e2 + 1.0f);
}

// ---------------------------------------------------------------------------
// k_cvt: conversions + state init + barrier reset
// ---------------------------------------------------------------------------
__global__ void k_cvt(const float* __restrict__ x, const float* __restrict__ h0,
                      const float* __restrict__ wih_fw, const float* __restrict__ whh_fw,
                      const float* __restrict__ wih_bw, const float* __restrict__ whh_bw,
                      _Float16* __restrict__ xf16,
                      _Float16* __restrict__ wihf16, _Float16* __restrict__ whhf16,
                      _Float16* __restrict__ wihb16, _Float16* __restrict__ whhb16,
                      _Float16* __restrict__ hf16, float* __restrict__ hf32,
                      int* __restrict__ bar) {
  size_t i0 = (size_t)blockIdx.x * blockDim.x + threadIdx.x;
  size_t stride = (size_t)gridDim.x * blockDim.x;
  for (size_t i = i0; i < XF_ELEMS; i += stride) xf16[i] = (_Float16)x[i];
  for (size_t i = i0; i < WEL; i += stride) {
    wihf16[i] = (_Float16)wih_fw[i];
    whhf16[i] = (_Float16)whh_fw[i];
    wihb16[i] = (_Float16)wih_bw[i];
    whhb16[i] = (_Float16)whh_bw[i];
  }
  for (size_t i = i0; i < (size_t)2 * B_ * H_; i += stride) {
    size_t dir = i / ((size_t)B_ * H_);
    size_t rem = i % ((size_t)B_ * H_);
    float v = h0[i];
    hf16[(dir * 2) * (size_t)B_ * H_ + rem] = (_Float16)v;   // parity 0
    hf32[(dir * 2) * (size_t)B_ * H_ + rem] = v;
  }
  if (i0 < 8) bar[i0] = 0;
}

// ---------------------------------------------------------------------------
// k_xgemm: xg[dir] = x @ w_ih[dir]^T   (raw, biases added in recurrence)
// ---------------------------------------------------------------------------
__global__ __launch_bounds__(256) void k_xgemm(const _Float16* __restrict__ xf16,
                                               const _Float16* __restrict__ wihf16,
                                               const _Float16* __restrict__ wihb16,
                                               float* __restrict__ xg_fw,
                                               float* __restrict__ xg_bw) {
  const int lane = threadIdx.x & 31;
  const int wave = threadIdx.x >> 5;
  const int lhalf = lane >> 4;
  const int l16 = lane & 15;
  const int dir = blockIdx.z;
  const _Float16* w = dir ? wihb16 : wihf16;
  float* xg = dir ? xg_bw : xg_fw;

  const int mtbase = blockIdx.x * 128 + wave * 16;
  const int nbase = blockIdx.y * 64;

  v8f acc[4];
#pragma unroll
  for (int j = 0; j < 4; ++j)
#pragma unroll
    for (int v = 0; v < 8; ++v) acc[j][v] = 0.0f;

  const _Float16* arow = xf16 + (size_t)(mtbase + l16) * I_;
  const _Float16* brow[4];
#pragma unroll
  for (int j = 0; j < 4; ++j)
    brow[j] = w + (size_t)(nbase + j * 16 + l16) * I_;

  for (int kk = 0; kk < I_; kk += 32) {
    v16h a = load_frag(arow + kk, lhalf);
#pragma unroll
    for (int j = 0; j < 4; ++j) {
      v16h b = load_frag(brow[j] + kk, lhalf);
      acc[j] = wmma16(a, b, acc[j]);
    }
  }
#pragma unroll
  for (int j = 0; j < 4; ++j)
#pragma unroll
    for (int v = 0; v < 8; ++v)
      xg[(size_t)(mtbase + v + 8 * lhalf) * G_ + (nbase + j * 16 + l16)] = acc[j][v];
}

// ---------------------------------------------------------------------------
// TDM: stage one 32-row x 1KB 2D tile (gate slice of w_hh, f16) into LDS.
// D# per CDNA5 ISA: data_size=3 (8B units) -> row =
// 128 units; tile 128x32; tensor dims/stride = 128/128/32. Groups 2/3 zero.
// ---------------------------------------------------------------------------
__device__ __forceinline__ void tdm_load_tile(const _Float16* gsrc,
                                              unsigned int lds_byte_off) {
#if __has_builtin(__builtin_amdgcn_tensor_load_to_lds)
  unsigned long long ga = (unsigned long long)(uintptr_t)gsrc;
  uint4v g0;
  g0[0] = 1u;                                          // count=1, user mode
  g0[1] = lds_byte_off;                                // lds_addr
  g0[2] = (unsigned int)ga;                            // global_addr[31:0]
  g0[3] = ((unsigned int)(ga >> 32) & 0x01FFFFFFu) | (2u << 30);  // [56:32]+type
  int8v g1;
  g1[0] = (3 << 16);        // workgroup_mask=0, data_size=3 (8B)
  g1[1] = (128 << 16);      // tensor_dim0[15:0]=128 (bits 63:48)
  g1[2] = (32 << 16);       // tensor_dim0 hi=0, tensor_dim1[15:0]=32
  g1[3] = (128 << 16);      // tensor_dim1 hi=0, tile_dim0=128
  g1[4] = 32;               // tile_dim1=32, tile_dim2=0
  g1[5] = 128;              // tensor_dim0_stride=128 units
  g1[6] = 0;
  g1[7] = 0;
  int4v gz = {0, 0, 0, 0};
#if __clang_major__ >= 23
  int8v gz8 = {0, 0, 0, 0, 0, 0, 0, 0};
  __builtin_amdgcn_tensor_load_to_lds(g0, g1, gz, gz, gz8, 0);
#else
  __builtin_amdgcn_tensor_load_to_lds(g0, g1, gz, gz, 0);
#endif
#else
  (void)gsrc; (void)lds_byte_off;
#endif
}

// ---------------------------------------------------------------------------
// k_recur: persistent recurrence. 64 blocks (32 per direction), 256 threads.
// ---------------------------------------------------------------------------
__global__ __launch_bounds__(256, 1) void k_recur(
    const float* __restrict__ xg_fw, const float* __restrict__ xg_bw,
    const _Float16* __restrict__ whhf16, const _Float16* __restrict__ whhb16,
    _Float16* __restrict__ hf16, float* __restrict__ hf32,
    const float* __restrict__ bih_fw, const float* __restrict__ bhh_fw,
    const float* __restrict__ bih_bw, const float* __restrict__ bhh_bw,
    float* __restrict__ out, int* __restrict__ bar) {
  const int tid = threadIdx.x;
  const int lane = tid & 31;
  const int wave = tid >> 5;
  const int lhalf = lane >> 4;
  const int l16 = lane & 15;

  const int dir = blockIdx.x >> 5;
  const int widx = blockIdx.x & 31;
  const int mhalf = widx >> 4;
  const int hblk = widx & 15;
  const int bbase = mhalf * 64;   // batch rows [bbase, bbase+64)
  const int cbase = hblk * 32;    // hidden cols [cbase, cbase+32)

  const float* xg = dir ? xg_bw : xg_fw;
  const _Float16* whh = dir ? whhb16 : whhf16;
  const float* bih = dir ? bih_bw : bih_fw;
  const float* bhh = dir ? bhh_bw : bhh_fw;

  __shared__ _Float16 wlds[3][32][H_];   // 96 KB
  __shared__ float gbuf[3][64][33];      // 25.3 KB

#if __has_builtin(__builtin_amdgcn_tensor_load_to_lds)
  // Tensor Data Mover: one descriptor per gate slice (32 rows x 1KB each).
  if (wave == 0) {
    const unsigned int ldsbase = (unsigned int)(uintptr_t)&wlds[0][0][0];
#pragma unroll
    for (int g = 0; g < 3; ++g)
      tdm_load_tile(whh + (size_t)(g * H_ + cbase) * H_,
                    ldsbase + (unsigned int)g * 32u * H_ * 2u);
    __builtin_amdgcn_s_wait_tensorcnt(0);
  }
#else
  for (int e = tid; e < 3 * 32 * (H_ / 8); e += 256) {
    int g = e / (32 * (H_ / 8));
    int rem = e % (32 * (H_ / 8));
    int c = rem / (H_ / 8);
    int k8 = rem % (H_ / 8);
    *(half8*)(&wlds[g][c][k8 * 8]) =
        *(const half8*)(whh + (size_t)(g * H_ + cbase + c) * H_ + k8 * 8);
  }
#endif
  __syncthreads();

  const int mt = wave & 3;
  const int ng = wave >> 2;
  const int arowbase = bbase + mt * 16 + l16;

  for (int s = 0; s < T_; ++s) {
    const int par = s & 1;
    const _Float16* hc = hf16 + (size_t)(dir * 2 + par) * B_ * H_;
    const float* hcf = hf32 + (size_t)(dir * 2 + par) * B_ * H_;
    _Float16* hn16 = hf16 + (size_t)(dir * 2 + (par ^ 1)) * B_ * H_;
    float* hn32 = hf32 + (size_t)(dir * 2 + (par ^ 1)) * B_ * H_;

    v8f acc[3];
#pragma unroll
    for (int i = 0; i < 3; ++i)
#pragma unroll
      for (int v = 0; v < 8; ++v) acc[i][v] = 0.0f;

    const _Float16* arow = hc + (size_t)arowbase * H_;
    for (int kk = 0; kk < H_; kk += 32) {
      v16h a = load_frag(arow + kk, lhalf);
#pragma unroll
      for (int i = 0; i < 3; ++i) {
        const int nt = ng * 3 + i;
        const int gate = nt >> 1;
        const int cb = nt & 1;
        v16h b = load_frag(&wlds[gate][cb * 16 + l16][kk], lhalf);
        acc[i] = wmma16(a, b, acc[i]);
      }
    }
#pragma unroll
    for (int i = 0; i < 3; ++i) {
      const int nt = ng * 3 + i;
      const int gate = nt >> 1;
      const int cb = nt & 1;
#pragma unroll
      for (int v = 0; v < 8; ++v)
        gbuf[gate][mt * 16 + v + 8 * lhalf][cb * 16 + l16] = acc[i][v];
    }
    __syncthreads();

    // Gate elementwise over this block's 64x32 slice of h'
    const int t = dir ? (T_ - 1 - s) : s;
    const int tn = dir ? (t - 1) : (t + 1);
    for (int e = tid; e < 64 * 32; e += 256) {
      const int m = e >> 5;
      const int c = e & 31;
      const int b = bbase + m;
      const int j = cbase + c;
      const size_t xo = ((size_t)t * B_ + b) * (size_t)G_;
      const float xr = xg[xo + j];
      const float xz = xg[xo + H_ + j];
      const float xn = xg[xo + 2 * H_ + j];
      if (s + 1 < T_) {   // prefetch next step's xg (streamed once from HBM)
        const size_t xon = ((size_t)tn * B_ + b) * (size_t)G_;
        __builtin_prefetch(xg + xon + j, 0, 3);
        __builtin_prefetch(xg + xon + H_ + j, 0, 3);
        __builtin_prefetch(xg + xon + 2 * H_ + j, 0, 3);
      }
      const float rp = xr + bih[j] + gbuf[0][m][c] + bhh[j];
      const float zp = xz + bih[H_ + j] + gbuf[1][m][c] + bhh[H_ + j];
      const float hnl = gbuf[2][m][c] + bhh[2 * H_ + j];
      const float r = sigmoid_f(rp);
      const float z = sigmoid_f(zp);
      const float n = tanh_f(xn + bih[2 * H_ + j] + r * hnl);
      const float hp = hcf[(size_t)b * H_ + j];
      float hv = (1.0f - z) * n + z * hp;
      hv = fminf(200.0f, fmaxf(-200.0f, hv));
      out[((size_t)t * B_ + b) * (size_t)(2 * H_) + dir * H_ + j] = hv;
      hn32[(size_t)b * H_ + j] = hv;
      hn16[(size_t)b * H_ + j] = (_Float16)hv;
      if (s == T_ - 1)
        out[(size_t)T_ * B_ * 2 * H_ + ((size_t)dir * B_ + b) * H_ + j] = hv;
    }
    __syncthreads();

    // Inter-workgroup step barrier (per direction)
    if (tid == 0) {
      __threadfence();
      atomicAdd(&bar[dir], 1);
      const int target = 32 * (s + 1);
      volatile int* vb = (volatile int*)&bar[dir];
      while (*vb < target) __builtin_amdgcn_s_sleep(1);
    }
    __syncthreads();
    __threadfence();
  }
}

// ---------------------------------------------------------------------------
extern "C" void kernel_launch(void* const* d_in, const int* in_sizes, int n_in,
                              void* d_out, int out_size, void* d_ws, size_t ws_size,
                              hipStream_t stream) {
  (void)in_sizes; (void)n_in; (void)out_size; (void)ws_size;
  const float* x      = (const float*)d_in[0];
  const float* h0     = (const float*)d_in[1];
  const float* wih_fw = (const float*)d_in[2];
  const float* whh_fw = (const float*)d_in[3];
  const float* bih_fw = (const float*)d_in[4];
  const float* bhh_fw = (const float*)d_in[5];
  const float* wih_bw = (const float*)d_in[6];
  const float* whh_bw = (const float*)d_in[7];
  const float* bih_bw = (const float*)d_in[8];
  const float* bhh_bw = (const float*)d_in[9];
  float* out = (float*)d_out;
  char* ws = (char*)d_ws;

  float*     xg_fw  = (float*)(ws + OFF_XG_FW);
  float*     xg_bw  = (float*)(ws + OFF_XG_BW);
  _Float16*  xf16   = (_Float16*)(ws + OFF_XF16);
  _Float16*  wihf16 = (_Float16*)(ws + OFF_WIHF);
  _Float16*  wihb16 = (_Float16*)(ws + OFF_WIHB);
  _Float16*  whhf16 = (_Float16*)(ws + OFF_WHHF);
  _Float16*  whhb16 = (_Float16*)(ws + OFF_WHHB);
  _Float16*  hf16   = (_Float16*)(ws + OFF_HF16);
  float*     hf32   = (float*)(ws + OFF_HF32);
  int*       bar    = (int*)(ws + OFF_BAR);

  k_cvt<<<2048, 256, 0, stream>>>(x, h0, wih_fw, whh_fw, wih_bw, whh_bw,
                                  xf16, wihf16, whhf16, wihb16, whhb16,
                                  hf16, hf32, bar);

  dim3 gg((T_ * B_) / 128, G_ / 64, 2);
  k_xgemm<<<gg, 256, 0, stream>>>(xf16, wihf16, wihb16, xg_fw, xg_bw);

  k_recur<<<64, 256, 0, stream>>>(xg_fw, xg_bw, whhf16, whhb16, hf16, hf32,
                                  bih_fw, bhh_fw, bih_bw, bhh_bw, out, bar);
}
